// HierarchicalEnvironmentDetector_32238024524302
// MI455X (gfx1250) — matrix operand order repaired and checked by reading, not compiled, not run on hardware
//
#include <hip/hip_runtime.h>
#include <hip/hip_bf16.h>
#include <math.h>
#include <stdint.h>

typedef __attribute__((ext_vector_type(16))) _Float16 v16h;
typedef __attribute__((ext_vector_type(8)))  _Float16 v8h;
typedef __attribute__((ext_vector_type(8)))  float    v8f;

#define B_    256
#define CIN   12
#define HIN   84
#define WIN   84
#define HID   1024
#define NC    16
#define NE    512
#define EPC   64
#define FLATF 3136   // 64*7*7

// ---------------------------------------------------------------------------
// CDNA5 async copy: 16 bytes per lane, global -> LDS, tracked by ASYNCcnt.
__device__ __forceinline__ void async_copy16(const void* gsrc, uint32_t ldsByteAddr) {
  asm volatile("global_load_async_to_lds_b128 %0, %1, off"
               :: "v"(ldsByteAddr), "v"((unsigned long long)(uintptr_t)gsrc)
               : "memory");
}
__device__ __forceinline__ void wait_async0() {
  asm volatile("s_wait_asynccnt 0x0" ::: "memory");
}

__device__ __forceinline__ v16h cat16(v8h lo, v8h hi) {
  v16h r;
#pragma unroll
  for (int i = 0; i < 8; ++i) { r[i] = lo[i]; r[i + 8] = hi[i]; }
  return r;
}

// A tile in LDS: f16 row-major [64][64] (KSTEP=64).  ISA 7.12.2 A-layout:
// lane L holds row M=smLocal+(L&15); elements 0..7 = K g*8..g*8+7,
// elements 8..15 = K 16+g*8.. -> two contiguous 16B LDS reads.
__device__ __forceinline__ v16h a_frag(const _Float16* ldsA, int smLocal, int s, int lane) {
  const int m  = smLocal + (lane & 15);
  const int g8 = (lane >> 4) * 8;
  const _Float16* row = ldsA + m * 64 + s * 32;
  const v8h lo = *(const v8h*)(row + g8);
  const v8h hi = *(const v8h*)(row + 16 + g8);
  return cat16(lo, hi);
}

// B tile in LDS: f16 stored N-major [TN][64] (pre-transposed weights).
// B-layout: lane = N; elements 0..7 = K g*8..g*8+7, elements 8..15 = K 16+g*8..
// (same K order as A) -> two 16B LDS reads.
__device__ __forceinline__ v16h b_frag(const _Float16* ldsBT, int snLocal, int s, int lane) {
  const int n  = snLocal + (lane & 15);
  const int k0 = (lane >> 4) << 3;
  const _Float16* row = ldsBT + n * 64 + s * 32 + k0;
  const v8h lo = *(const v8h*)(row);
  const v8h hi = *(const v8h*)(row + 16);
  return cat16(lo, hi);
}

// ---------------------------------------------------------------------------
// Tiled WMMA GEMM on f16 operands (A row-major MxK, BT N-major NxK).
// Block = 256 threads (8 waves), C tile = 64 x TN, K-step 64,
// double-buffered async LDS staging.
//   EPI=0 : val = relu(acc+bias); optionally store f32 C and/or f16 C16
//   EPI=1 : atomicAdd(C[m, mapping[n]], probs[m,c] * sigmoid(acc+bias))
template <int TN, int EPI>
__global__ void gemm_wmma(const _Float16* __restrict__ A, long aStride,
                          const _Float16* __restrict__ BT, long bStride,
                          const float* __restrict__ bias, long biasStride,
                          float* __restrict__ C, int ldc, long cStride,
                          _Float16* __restrict__ C16, int ldc16, long cStride16,
                          int K,
                          const float* __restrict__ probs,
                          const int* __restrict__ mapping) {
  constexpr int SUB_COLS = TN / 16;          // 16x16 subtile columns (4 or 8)
  constexpr int WPC      = 8 / SUB_COLS;     // waves per subtile column
  constexpr int RPW      = 4 / WPC;          // row-subtiles per wave (4 or 2)

  __shared__ __align__(16) _Float16 ldsA[2][64 * 64];
  __shared__ __align__(16) _Float16 ldsBT[2][TN * 64];

  const int tid   = threadIdx.x;
  const int lane  = tid & 31;
  const int wave  = tid >> 5;
  const int mBase = blockIdx.x * 64;
  const int nBase = blockIdx.y * TN;
  const int c     = blockIdx.z;

  const int col  = wave % SUB_COLS;
  const int row0 = (wave / SUB_COLS) * RPW;

  const _Float16* Ap = A + (size_t)c * aStride;
  const _Float16* Bp = BT + (size_t)c * bStride;

  const uint32_t aLds0 = (uint32_t)(uintptr_t)&ldsA[0][0];
  const uint32_t aLds1 = (uint32_t)(uintptr_t)&ldsA[1][0];
  const uint32_t bLds0 = (uint32_t)(uintptr_t)&ldsBT[0][0];
  const uint32_t bLds1 = (uint32_t)(uintptr_t)&ldsBT[1][0];

  // stage one K-tile: A 64x64 halves (8KB), BT TNx64 halves
  auto stage = [&](int kb, int buf) {
    const uint32_t aL = buf ? aLds1 : aLds0;
    const uint32_t bL = buf ? bLds1 : bLds0;
#pragma unroll
    for (int ch = 0; ch < 2; ++ch) {               // 512 chunks of 16B
      const int cidx = tid + ch * 256;
      const int r    = cidx >> 3;                  // row 0..63
      const int cc   = (cidx & 7) * 8;             // half offset in row
      async_copy16(Ap + (size_t)(mBase + r) * K + kb + cc, aL + cidx * 16);
    }
#pragma unroll
    for (int ch = 0; ch < TN * 8 / 256; ++ch) {    // TN*8 chunks of 16B
      const int cidx = tid + ch * 256;
      const int r    = cidx >> 3;                  // n-row 0..TN-1
      const int cc   = (cidx & 7) * 8;
      async_copy16(Bp + (size_t)(nBase + r) * K + kb + cc, bL + cidx * 16);
    }
  };

  v8f acc[RPW];
#pragma unroll
  for (int r = 0; r < RPW; ++r) acc[r] = v8f{};

  stage(0, 0);
  wait_async0();
  __syncthreads();

  int buf = 0;
  for (int kb = 0; kb < K; kb += 64) {
    if (kb + 64 < K) stage(kb + 64, buf ^ 1);      // overlap copy with math

    const _Float16* curA = &ldsA[buf][0];
    const _Float16* curB = &ldsBT[buf][0];
#pragma unroll
    for (int s = 0; s < 2; ++s) {                  // two k=32 slices
      const v16h bf = b_frag(curB, col * 16, s, lane);
#pragma unroll
      for (int r = 0; r < RPW; ++r) {
        const v16h af = a_frag(curA, (row0 + r) * 16, s, lane);
        acc[r] = __builtin_amdgcn_wmma_f32_16x16x32_f16(false, af, false, bf,
                                                        (short)0, acc[r], false, false);
      }
    }
    wait_async0();
    __syncthreads();
    buf ^= 1;
  }

  // ---- epilogue -----------------------------------------------------------
  const int n  = nBase + col * 16 + (lane & 15);
  const int mo = (lane >> 4) << 3;
  const float bv = bias[(size_t)c * biasStride + n];
  if (EPI == 0) {
    float*     Cp = C   ? C   + (size_t)c * cStride   : nullptr;
    _Float16*  Hp = C16 ? C16 + (size_t)c * cStride16 : nullptr;
#pragma unroll
    for (int r = 0; r < RPW; ++r) {
      const int mrow = mBase + (row0 + r) * 16 + mo;
#pragma unroll
      for (int q = 0; q < 8; ++q) {
        const float val = fmaxf(acc[r][q] + bv, 0.f);
        if (Cp) Cp[(size_t)(mrow + q) * ldc + n] = val;
        if (Hp) Hp[(size_t)(mrow + q) * ldc16 + n] = (_Float16)val;
      }
    }
  } else {
    const int tgt = mapping[c * EPC + n];          // nBase==0 for EPI=1
    float* Cp = C;
#pragma unroll
    for (int r = 0; r < RPW; ++r) {
      const int mrow = mBase + (row0 + r) * 16 + mo;
#pragma unroll
      for (int q = 0; q < 8; ++q) {
        const int m = mrow + q;
        const float x   = acc[r][q] + bv;
        const float sig = 1.f / (1.f + __expf(-x));
        atomicAdd(&Cp[(size_t)m * NE + tgt], probs[m * NC + c] * sig);
      }
    }
  }
}

// one-time operand prep: out[z][n*K+k] = (f16) in[z][k*N+n]
__global__ void cvt_transpose_f16(const float* __restrict__ in,
                                  _Float16* __restrict__ out, int K, int N) {
  const size_t zofs = (size_t)blockIdx.z * K * N;
  const int idx = blockIdx.x * blockDim.x + threadIdx.x;
  if (idx >= K * N) return;
  const int k = idx / N, n = idx % N;
  out[zofs + (size_t)n * K + k] = (_Float16)in[zofs + idx];
}

// ---------------- direct convolutions (VALU; small share of total work) ------
__global__ void conv1_relu(const float* __restrict__ x, const float* __restrict__ k,
                           const float* __restrict__ bias, float* __restrict__ y) {
  int idx = blockIdx.x * blockDim.x + threadIdx.x;
  if (idx >= B_ * 32 * 400) return;
  const int w = idx % 20, h = (idx / 20) % 20, oc = (idx / 400) % 32,
            n = idx / (400 * 32);
  float acc = bias[oc];
  const float* xin = x + (size_t)n * CIN * HIN * WIN;
  const float* kk  = k + (size_t)oc * CIN * 64;
  for (int ic = 0; ic < CIN; ++ic)
#pragma unroll
    for (int kh = 0; kh < 8; ++kh)
#pragma unroll
      for (int kw = 0; kw < 8; ++kw)
        acc = fmaf(xin[ic * HIN * WIN + (h * 4 + kh) * WIN + (w * 4 + kw)],
                   kk[ic * 64 + kh * 8 + kw], acc);
  y[idx] = fmaxf(acc, 0.f);
}

__global__ void conv2_relu(const float* __restrict__ x, const float* __restrict__ k,
                           const float* __restrict__ bias, float* __restrict__ y) {
  int idx = blockIdx.x * blockDim.x + threadIdx.x;
  if (idx >= B_ * 64 * 81) return;
  const int w = idx % 9, h = (idx / 9) % 9, oc = (idx / 81) % 64,
            n = idx / (81 * 64);
  float acc = bias[oc];
  const float* xin = x + (size_t)n * 32 * 400;
  const float* kk  = k + (size_t)oc * 32 * 16;
  for (int ic = 0; ic < 32; ++ic)
#pragma unroll
    for (int kh = 0; kh < 4; ++kh)
#pragma unroll
      for (int kw = 0; kw < 4; ++kw)
        acc = fmaf(xin[ic * 400 + (h * 2 + kh) * 20 + (w * 2 + kw)],
                   kk[ic * 16 + kh * 4 + kw], acc);
  y[idx] = fmaxf(acc, 0.f);
}

// conv3 feeds the proj GEMM only -> emit f16 feats directly
__global__ void conv3_relu_f16(const float* __restrict__ x, const float* __restrict__ k,
                               const float* __restrict__ bias, _Float16* __restrict__ y) {
  int idx = blockIdx.x * blockDim.x + threadIdx.x;
  if (idx >= B_ * FLATF) return;
  const int w = idx % 7, h = (idx / 7) % 7, oc = (idx / 49) % 64,
            n = idx / FLATF;
  float acc = bias[oc];
  const float* xin = x + (size_t)n * 64 * 81;
  const float* kk  = k + (size_t)oc * 64 * 9;
  for (int ic = 0; ic < 64; ++ic)
#pragma unroll
    for (int kh = 0; kh < 3; ++kh)
#pragma unroll
      for (int kw = 0; kw < 3; ++kw)
        acc = fmaf(xin[ic * 81 + (h + kh) * 9 + (w + kw)],
                   kk[ic * 9 + kh * 3 + kw], acc);
  y[idx] = (_Float16)fmaxf(acc, 0.f);               // feats row-major B x 3136
}

// category head: M=256, N=16, K=1024, f16 operands (WcatT is [16][1024]).
__global__ void gemm_cat(const _Float16* __restrict__ A, const _Float16* __restrict__ BT,
                         const float* __restrict__ bias, float* __restrict__ C) {
  const int lane  = threadIdx.x & 31;
  const int mBase = blockIdx.x * 16;
  v8f acc = {};
  const int m  = mBase + (lane & 15);
  const int g8 = (lane >> 4) * 8;
  const int n  = lane & 15;
  const int k0 = (lane >> 4) << 4;
  for (int kb = 0; kb < HID; kb += 32) {
    const _Float16* arow = A + (size_t)m * HID + kb;
    const v16h a = cat16(*(const v8h*)(arow + g8), *(const v8h*)(arow + 16 + g8));
    const _Float16* brow = BT + (size_t)n * HID + kb + g8;
    const v16h b = cat16(*(const v8h*)(brow), *(const v8h*)(brow + 16));
    acc = __builtin_amdgcn_wmma_f32_16x16x32_f16(false, a, false, b,
                                                 (short)0, acc, false, false);
  }
  const int m0 = mBase + ((lane >> 4) << 3);
  const float bv = bias[n];
#pragma unroll
  for (int r = 0; r < 8; ++r)
    C[(size_t)(m0 + r) * NC + n] = acc[r] + bv;
}

__global__ void softmax16(const float* __restrict__ logits, float* __restrict__ probs) {
  const int b = blockIdx.x * blockDim.x + threadIdx.x;
  if (b >= B_) return;
  float v[NC], mx = -1e30f;
#pragma unroll
  for (int c = 0; c < NC; ++c) { v[c] = logits[b * NC + c]; mx = fmaxf(mx, v[c]); }
  float s = 0.f;
#pragma unroll
  for (int c = 0; c < NC; ++c) { v[c] = __expf(v[c] - mx); s += v[c]; }
  const float inv = 1.f / s;
#pragma unroll
  for (int c = 0; c < NC; ++c) probs[b * NC + c] = v[c] * inv;
}

__global__ void zero_f32(float* __restrict__ p, int n) {
  int i = blockIdx.x * blockDim.x + threadIdx.x;
  if (i < n) p[i] = 0.f;
}

// ---------------------------------------------------------------------------
extern "C" void kernel_launch(void* const* d_in, const int* in_sizes, int n_in,
                              void* d_out, int out_size, void* d_ws, size_t ws_size,
                              hipStream_t stream) {
  (void)in_sizes; (void)n_in; (void)out_size; (void)ws_size;
  const float* obs   = (const float*)d_in[0];
  const float* k1    = (const float*)d_in[1];
  const float* cb1   = (const float*)d_in[2];
  const float* k2    = (const float*)d_in[3];
  const float* cb2   = (const float*)d_in[4];
  const float* k3    = (const float*)d_in[5];
  const float* cb3   = (const float*)d_in[6];
  const float* Wproj = (const float*)d_in[7];
  const float* bproj = (const float*)d_in[8];
  const float* Wcat  = (const float*)d_in[9];
  const float* bcat  = (const float*)d_in[10];
  const float* W1    = (const float*)d_in[11];
  const float* b1    = (const float*)d_in[12];
  const float* W2    = (const float*)d_in[13];
  const float* b2    = (const float*)d_in[14];
  const int*   mapping = (const int*)d_in[15];

  float* out    = (float*)d_out;
  float* logits = out;                          // [256,16]
  float* expert = out + (size_t)B_ * NC;        // [256,512]
  float* hidden = expert + (size_t)B_ * NE;     // [256,1024] (f32 output)

  // workspace carve-up: f32 regions first, then f16 regions (16B aligned)
  float* ws     = (float*)d_ws;
  float* y1     = ws;                                 // [256,32,20,20]
  float* y2     = y1 + (size_t)B_ * 32 * 400;         // [256,64,9,9]
  float* probs  = y2 + (size_t)B_ * 64 * 81;          // [256,16]
  _Float16* featsH  = (_Float16*)(probs + B_ * NC);   // [256,3136]
  _Float16* hiddenH = featsH + (size_t)B_ * FLATF;    // [256,1024]
  _Float16* h1H     = hiddenH + (size_t)B_ * HID;     // [16,256,512]
  _Float16* WprojT  = h1H + (size_t)NC * B_ * (HID / 2);   // [1024,3136]
  _Float16* W1T     = WprojT + (size_t)FLATF * HID;        // [16,512,1024]
  _Float16* W2T     = W1T + (size_t)NC * HID * (HID / 2);  // [16,64,512]
  _Float16* WcatT   = W2T + (size_t)NC * EPC * (HID / 2);  // [16,1024]

  // ---- one-time operand prep: f32 -> f16, N-major transpose ---------------
  cvt_transpose_f16<<<dim3((FLATF * HID + 255) / 256, 1, 1), 256, 0, stream>>>(
      Wproj, WprojT, FLATF, HID);
  cvt_transpose_f16<<<dim3((HID * (HID / 2) + 255) / 256, 1, NC), 256, 0, stream>>>(
      W1, W1T, HID, HID / 2);
  cvt_transpose_f16<<<dim3(((HID / 2) * EPC + 255) / 256, 1, NC), 256, 0, stream>>>(
      W2, W2T, HID / 2, EPC);
  cvt_transpose_f16<<<dim3((HID * NC + 255) / 256, 1, 1), 256, 0, stream>>>(
      Wcat, WcatT, HID, NC);

  // ---- encoder ------------------------------------------------------------
  conv1_relu<<<(B_ * 32 * 400 + 255) / 256, 256, 0, stream>>>(obs, k1, cb1, y1);
  conv2_relu<<<(B_ * 64 * 81 + 255) / 256, 256, 0, stream>>>(y1, k2, cb2, y2);
  conv3_relu_f16<<<(B_ * FLATF + 255) / 256, 256, 0, stream>>>(y2, k3, cb3, featsH);

  // hidden = relu(feats @ Wproj + bproj): f32 to d_out + f16 operand copy
  gemm_wmma<128, 0><<<dim3(B_ / 64, HID / 128, 1), 256, 0, stream>>>(
      featsH, 0, WprojT, 0, bproj, 0,
      hidden, HID, 0, hiddenH, HID, 0, FLATF, nullptr, nullptr);

  // category logits + softmax probs
  gemm_cat<<<B_ / 16, 32, 0, stream>>>(hiddenH, WcatT, bcat, logits);
  softmax16<<<(B_ + 63) / 64, 64, 0, stream>>>(logits, probs);

  // zero expert accumulation buffer (d_out is poisoned)
  zero_f32<<<(B_ * NE + 255) / 256, 256, 0, stream>>>(expert, B_ * NE);

  // h1[c] = relu(hidden @ W1[c] + b1[c])  (f16-only result)
  gemm_wmma<128, 0><<<dim3(B_ / 64, (HID / 2) / 128, NC), 256, 0, stream>>>(
      hiddenH, 0, W1T, (long)(HID / 2) * HID, b1, HID / 2,
      nullptr, 0, 0, h1H, HID / 2, (long)B_ * (HID / 2), HID, nullptr, nullptr);

  // weighted sigmoid + scatter-add into expert buffer
  gemm_wmma<64, 1><<<dim3(B_ / 64, 1, NC), 256, 0, stream>>>(
      h1H, (long)B_ * (HID / 2), W2T, (long)EPC * (HID / 2), b2, EPC,
      expert, 0, 0, nullptr, 0, 0, HID / 2, probs, mapping);
}
